// Aligner_28458453303304
// MI455X (gfx1250) — matrix operand order repaired and checked by reading, not compiled, not run on hardware
//
#include <hip/hip_runtime.h>
#include <cstdint>
#include <cstddef>

#define BB 16
#define CC 64
#define TY 4096
#define TX 512
#define KK 128
#define NEGV -1.0e9f

// ---------- WMMA types (CDNA5 gfx1250, wave32) ----------
typedef __attribute__((ext_vector_type(16))) __bf16 v16bf;
typedef __attribute__((ext_vector_type(8)))  float  v8f;

struct alignas(16) U4 { unsigned x, y, z, w; };
struct alignas(32) U8 { U4 lo, hi; };

static __device__ inline v16bf make_v16bf(U4 lo, U4 hi) {
  U8 p{lo, hi};
  return __builtin_bit_cast(v16bf, p);
}

static __device__ inline unsigned short f2bf(float f) {
  unsigned u = __float_as_uint(f);
  unsigned r = u + 0x7FFFu + ((u >> 16) & 1u);   // round-to-nearest-even
  return (unsigned short)(r >> 16);
}

// ---------- workspace layout (bytes) ----------
// meta: int[0..15]=t_x_len, int[16..31]=t_y_len; float idx 32=logdet_sum, 33=ym_sum, 34=kl_sum
static const size_t OFF_A     = 1024;                         // bf16 A  [B][TY][KK]
static const size_t OFF_BT    = OFF_A   + (size_t)BB*TY*KK*2; // bf16 B^T [B][TX][KK]
static const size_t OFF_TCD   = OFF_BT  + (size_t)BB*TX*KK*2; // f32 termC+termD [B][TX]
static const size_t OFF_SC    = OFF_TCD + (size_t)BB*TX*4;    // f32 scores [B][TY][TX]
static const size_t OFF_IDX   = OFF_SC  + (size_t)BB*TY*TX*4; // i32 path index [B][TY]
static const size_t OFF_START = OFF_IDX + (size_t)BB*TY*4;    // i32 seg start [B][TX]
static const size_t OFF_CNT   = OFF_START + (size_t)BB*TX*4;  // i32 seg count [B][TX]

// ---------------- kernel 0: lengths / scalar sums / zero accumulators ----------------
__global__ void k0_meta(const unsigned char* __restrict__ xm,
                        const unsigned char* __restrict__ ym,
                        const float* __restrict__ logdet,
                        int* meta_i, float* meta_f) {
  int t = threadIdx.x;
  if (t < BB) {
    int sx = 0;
    for (int x = 0; x < TX; x++) sx += xm[t * TX + x] ? 1 : 0;
    int sy = 0;
    for (int y = 0; y < TY; y++) sy += ym[t * TY + y] ? 1 : 0;
    meta_i[t] = sx;
    meta_i[BB + t] = sy;
  }
  __syncthreads();
  if (t == 0) {
    float ld = 0.f, ys = 0.f;
    for (int b = 0; b < BB; b++) { ld += logdet[b]; ys += (float)meta_i[BB + b]; }
    meta_f[32] = ld;
    meta_f[33] = ys;
    meta_f[34] = 0.f;   // kl accumulator
  }
}

// ---------------- prep A: A[y][c]=bf(z^2), A[y][64+c]=bf(z)  (transpose z[C,TY] via LDS) ----
__global__ void kprep_a(const float* __restrict__ zf, unsigned short* __restrict__ Abf) {
  __shared__ float tile[CC * 65];
  int b = blockIdx.y;
  int y0 = blockIdx.x * 64;
  int t = threadIdx.x;
  {
    int yl = t & 63, cq = t >> 6;
    for (int cc = 0; cc < 16; cc++) {
      int c = cq * 16 + cc;
      tile[c * 65 + yl] = zf[((size_t)(b * CC + c)) * TY + y0 + yl];
    }
  }
  __syncthreads();
  {
    int c = t & 63, yq = t >> 6;
    for (int j = 0; j < 16; j++) {
      int yl = yq * 16 + j;
      float z = tile[c * 65 + yl];
      size_t row = ((size_t)b * TY + y0 + yl) * KK;
      Abf[row + c]      = f2bf(z * z);
      Abf[row + 64 + c] = f2bf(z);
    }
  }
}

// ---------------- prep B: BT[x][c]=bf(o), BT[x][64+c]=bf(-2*m*o); tcd[x]=sum_c(m^2*o+2*l) ----
__global__ void kprep_b(const float* __restrict__ mp, const float* __restrict__ lp,
                        unsigned short* __restrict__ BT, float* __restrict__ tcd) {
  __shared__ float tO[CC * 65];
  __shared__ float tM[CC * 65];
  __shared__ float tS[4 * 64];
  int b = blockIdx.y;
  int x0 = blockIdx.x * 64;
  int t = threadIdx.x;
  {
    int xl = t & 63, cq = t >> 6;
    float acc = 0.f;
    for (int cc = 0; cc < 16; cc++) {
      int c = cq * 16 + cc;
      float m = mp[((size_t)(b * CC + c)) * TX + x0 + xl];
      float l = lp[((size_t)(b * CC + c)) * TX + x0 + xl];
      float o = __expf(-2.f * l);
      tO[c * 65 + xl] = o;
      tM[c * 65 + xl] = -2.f * m * o;
      acc += m * m * o + 2.f * l;
    }
    tS[cq * 64 + xl] = acc;
  }
  __syncthreads();
  {
    int c = t & 63, xq = t >> 6;
    for (int j = 0; j < 16; j++) {
      int xi = xq * 16 + j;
      size_t row = ((size_t)b * TX + x0 + xi) * KK;
      BT[row + c]      = f2bf(tO[c * 65 + xi]);
      BT[row + 64 + c] = f2bf(tM[c * 65 + xi]);
    }
  }
  if (t < 64)
    tcd[b * TX + x0 + t] = tS[t] + tS[64 + t] + tS[128 + t] + tS[192 + t];
}

// ---------------- WMMA GEMM: scores[b][y][x] = -0.5*(A·B + tcd[x]) --------------------------
// Per wave: one 16(M=y) x 64(N=x) tile, K=128 in four 16x16x32 bf16 WMMAs per 16-col subtile.
__global__ void kgemm(const unsigned short* __restrict__ Abf,
                      const unsigned short* __restrict__ BT,
                      const float* __restrict__ tcd,
                      float* __restrict__ sc) {
  int b   = blockIdx.z;
  int yb  = blockIdx.y;           // 0..31  -> 128 rows each
  int xb  = blockIdx.x;           // 0..7   -> 64 cols each
  int t   = threadIdx.x;
  int wave = t >> 5, lane = t & 31;
  int m = lane & 15, h = lane >> 4;
  int y0 = yb * 128 + wave * 16;
  int x0 = xb * 64;

  const unsigned short* Arow = Abf + ((size_t)b * TY + y0 + m) * KK;

  v8f zero = {0.f, 0.f, 0.f, 0.f, 0.f, 0.f, 0.f, 0.f};
  v8f acc[4];
#pragma unroll
  for (int i = 0; i < 4; i++) acc[i] = zero;

#pragma unroll
  for (int kc = 0; kc < 4; kc++) {
    // 16-bit A 16x32 layout: VGPR0-3 = K = h*8+0..7 ; VGPR4-7 = K = 16+h*8+0..7 (per lane row m)
    U4 alo = *(const U4*)(Arow + kc * 32 + h * 8);
    U4 ahi = *(const U4*)(Arow + kc * 32 + 16 + h * 8);
    v16bf av = make_v16bf(alo, ahi);
#pragma unroll
    for (int xt = 0; xt < 4; xt++) {
      // B 32x16 layout: lane = col n (=m), VGPR0-3 = K = h*8+0..7 ; VGPR4-7 = K = 16+h*8+0..7
      const unsigned short* Brow =
          BT + ((size_t)b * TX + x0 + xt * 16 + m) * KK + kc * 32 + h * 8;
      U4 blo = *(const U4*)(Brow);
      U4 bhi = *(const U4*)(Brow + 16);
      v16bf bv = make_v16bf(blo, bhi);
      acc[xt] = __builtin_amdgcn_wmma_f32_16x16x32_bf16(
          false, av, false, bv, (short)0, acc[xt], false, false);
    }
  }

  // C/D layout: VGPR r -> row m = h*8 + r, col n = lane&15
#pragma unroll
  for (int xt = 0; xt < 4; xt++) {
    int xc = x0 + xt * 16 + m;
    float tc = tcd[b * TX + xc];
#pragma unroll
    for (int r = 0; r < 8; r++) {
      int yy = y0 + h * 8 + r;
      sc[((size_t)b * TY + yy) * TX + xc] = -0.5f * (acc[xt][r] + tc);
    }
  }
}

// ---------------- fused forward+backward MAS DP: one wave32 per batch ------------------------
static __device__ inline void load_row16(const float* __restrict__ rowp, float* v) {
  const float4* p = (const float4*)rowp;
  float4 q0 = p[0], q1 = p[1], q2 = p[2], q3 = p[3];
  v[0] = q0.x; v[1] = q0.y; v[2]  = q0.z; v[3]  = q0.w;
  v[4] = q1.x; v[5] = q1.y; v[6]  = q1.z; v[7]  = q1.w;
  v[8] = q2.x; v[9] = q2.y; v[10] = q2.z; v[11] = q2.w;
  v[12] = q3.x; v[13] = q3.y; v[14] = q3.z; v[15] = q3.w;
}

__global__ void kdp(const float* __restrict__ sc, const int* __restrict__ meta_i,
                    int* __restrict__ idxOut) {
  extern __shared__ unsigned short cmpB[];   // [TY][32] decision bits, 256 KB
  int b = blockIdx.x;
  int lane = threadIdx.x;                    // wave32: x = lane*16 + i
  int t_x = meta_i[b], t_y = meta_i[BB + b];
  const float* S = sc + (size_t)b * TY * TX;

  float prev[16], cur[16], nxt[16];
#pragma unroll
  for (int i = 0; i < 16; i++) prev[i] = NEGV;
  load_row16(S + lane * 16, cur);

  for (int y = 0; y < TY; y++) {
    if (y + 1 < TY) load_row16(S + (size_t)(y + 1) * TX + lane * 16, nxt);
    if (y + 8 < TY) __builtin_prefetch(S + (size_t)(y + 8) * TX + lane * 16, 0, 1);

    float carry = __shfl_up(prev[15], 1);    // lane-1's last element
    int lo = t_x + y - t_y; if (lo < 0) lo = 0;
    int hi = (t_x < y + 1) ? t_x : (y + 1);

    unsigned bits = 0;
    float val[16];
#pragma unroll
    for (int i = 0; i < 16; i++) {
      int x = (lane << 4) + i;
      float pl   = (i == 0) ? carry : prev[i - 1];
      float left = (x == 0) ? ((y == 0) ? 0.f : NEGV) : pl;   // vals[y-1][x-1] w/ diag seed
      float bl   = (x == 0) ? prev[0] : pl;                   // clamp for backward compare
      bits |= (prev[i] < bl ? 1u : 0u) << i;                  // v_at < v_left on row y-1
      float vc = (x == y) ? NEGV : prev[i];
      float v  = fmaxf(vc, left) + cur[i];
      val[i]   = (x >= lo && x < hi) ? v : NEGV;
    }
    cmpB[y * 32 + lane] = (unsigned short)bits;
#pragma unroll
    for (int i = 0; i < 16; i++) { prev[i] = val[i]; cur[i] = nxt[i]; }
  }
  __syncthreads();

  if (lane == 0) {                           // backward walk, purely from LDS bits
    int index = t_x - 1;
    for (int y = TY - 1; y >= 0; y--) {
      bool active = y < t_y;
      idxOut[b * TY + y] = active ? index : 0x40000000;   // sentinel keeps array sorted
      if (active && index != 0) {
        bool dec = (index == y) ||
                   (((cmpB[y * 32 + (index >> 4)] >> (index & 15)) & 1u) != 0u);
        if (dec) index--;
      }
    }
  }
}

// ---------------- durations / segment ranges / pad_mask --------------------------------------
__global__ void kdur(const int* __restrict__ idx, const unsigned char* __restrict__ xm,
                     int* __restrict__ starts, int* __restrict__ cnts,
                     float* __restrict__ dur_out, float* __restrict__ pad_out) {
  __shared__ int s_idx[TY];
  int b = blockIdx.x, t = threadIdx.x;
  for (int j = t; j < TY; j += 512) s_idx[j] = idx[b * TY + j];
  __syncthreads();
  int x = t;
  int lo = 0, hi = TY;
  while (lo < hi) { int mid = (lo + hi) >> 1; if (s_idx[mid] < x) lo = mid + 1; else hi = mid; }
  int l0 = lo;
  lo = 0; hi = TY;
  while (lo < hi) { int mid = (lo + hi) >> 1; if (s_idx[mid] <= x) lo = mid + 1; else hi = mid; }
  int cnt = lo - l0;
  starts[b * TX + x] = l0;
  cnts[b * TX + x]   = cnt;
  dur_out[b * TX + x] = (float)cnt;
  pad_out[b * TX + x] = xm[b * TX + x] ? 0.f : 1.f;
}

// ---------------- z_pooled[b][x][c] = sum(z_spec[b][c][seg]) / t_x_len -----------------------
__global__ void kpool(const float* __restrict__ zs, const int* __restrict__ starts,
                      const int* __restrict__ cnts, const int* __restrict__ meta_i,
                      float* __restrict__ zp_out) {
  int b = blockIdx.y;
  int x = blockIdx.x * 4 + (threadIdx.x >> 6);
  int c = threadIdx.x & 63;
  int s = starts[b * TX + x], n = cnts[b * TX + x];
  float inv = 1.f / (float)meta_i[b];
  const float* zrow = zs + ((size_t)(b * CC + c)) * TY;
  float acc = 0.f;
  for (int j = 0; j < n; j++) acc += zrow[s + j];
  zp_out[((size_t)b * TX + x) * CC + c] = acc * inv;
}

// ---------------- KL reduction over (b,c,y) with gathered m_p/logs_p -------------------------
__global__ void kkl(const float* __restrict__ zf, const float* __restrict__ mp,
                    const float* __restrict__ lp, const int* __restrict__ idx,
                    const int* __restrict__ meta_i, float* meta_f) {
  __shared__ float red[256];
  int b = blockIdx.y;
  int y = blockIdx.x * 256 + threadIdx.x;
  int t = threadIdx.x;
  int t_y = meta_i[BB + b];
  float acc = 0.f;
  if (y < t_y) {
    int ix = idx[b * TY + y];
    for (int c = 0; c < CC; c++) {
      float m = mp[((size_t)(b * CC + c)) * TX + ix];
      float l = lp[((size_t)(b * CC + c)) * TX + ix];
      float z = zf[((size_t)(b * CC + c)) * TY + y];
      float d = z - m;
      acc += l + 0.5f * __expf(-2.f * l) * d * d;
    }
  }
  red[t] = acc;
  __syncthreads();
  for (int s = 128; s > 0; s >>= 1) { if (t < s) red[t] += red[t + s]; __syncthreads(); }
  if (t == 0) atomicAdd(&meta_f[34], red[0]);
}

__global__ void kfin(const float* meta_f, float* loss_out) {
  if (threadIdx.x == 0)
    loss_out[0] = (meta_f[34] - meta_f[32]) / meta_f[33];
}

// ---------------- host launcher ----------------
extern "C" void kernel_launch(void* const* d_in, const int* in_sizes, int n_in,
                              void* d_out, int out_size, void* d_ws, size_t ws_size,
                              hipStream_t stream) {
  const float* z_spec = (const float*)d_in[0];   // [B,C,TY]
  const float* z_flow = (const float*)d_in[1];   // [B,C,TY]
  const float* logdet = (const float*)d_in[2];   // [B]
  const float* m_p    = (const float*)d_in[3];   // [B,C,TX]
  const float* logs_p = (const float*)d_in[4];   // [B,C,TX]
  const unsigned char* x_mask = (const unsigned char*)d_in[5]; // bool [B,1,TX]
  const unsigned char* y_mask = (const unsigned char*)d_in[6]; // bool [B,1,TY]

  char* ws = (char*)d_ws;
  int*   meta_i = (int*)ws;
  float* meta_f = (float*)ws;
  unsigned short* Abf = (unsigned short*)(ws + OFF_A);
  unsigned short* BT  = (unsigned short*)(ws + OFF_BT);
  float* tcd    = (float*)(ws + OFF_TCD);
  float* sc     = (float*)(ws + OFF_SC);
  int*   idxw   = (int*)(ws + OFF_IDX);
  int*   starts = (int*)(ws + OFF_START);
  int*   cnts   = (int*)(ws + OFF_CNT);

  float* zp_out   = (float*)d_out;                 // [B,TX,C]
  float* dur_out  = zp_out + (size_t)BB * TX * CC; // [B,TX]
  float* loss_out = dur_out + (size_t)BB * TX;     // [1]
  float* pad_out  = loss_out + 1;                  // [B,TX]

  hipLaunchKernelGGL(k0_meta, dim3(1), dim3(256), 0, stream,
                     x_mask, y_mask, logdet, meta_i, meta_f);
  hipLaunchKernelGGL(kprep_a, dim3(TY / 64, BB), dim3(256), 0, stream, z_flow, Abf);
  hipLaunchKernelGGL(kprep_b, dim3(TX / 64, BB), dim3(256), 0, stream, m_p, logs_p, BT, tcd);
  hipLaunchKernelGGL(kgemm, dim3(TX / 64, TY / 128, BB), dim3(256), 0, stream,
                     Abf, BT, tcd, sc);
  hipLaunchKernelGGL(kdp, dim3(BB), dim3(32), TY * 32 * sizeof(unsigned short), stream,
                     sc, meta_i, idxw);
  hipLaunchKernelGGL(kdur, dim3(BB), dim3(512), 0, stream,
                     idxw, x_mask, starts, cnts, dur_out, pad_out);
  hipLaunchKernelGGL(kpool, dim3(TX / 4, BB), dim3(256), 0, stream,
                     z_spec, starts, cnts, meta_i, zp_out);
  hipLaunchKernelGGL(kkl, dim3(TY / 256, BB), dim3(256), 0, stream,
                     z_flow, m_p, logs_p, idxw, meta_i, meta_f);
  hipLaunchKernelGGL(kfin, dim3(1), dim3(32), 0, stream, meta_f, loss_out);
  (void)in_sizes; (void)n_in; (void)out_size; (void)ws_size;
}
